// ASGCN_BERT_Unit_45896020525363
// MI455X (gfx1250) — matrix operand order repaired and checked by reading, not compiled, not run on hardware
//
#include <hip/hip_runtime.h>
#include <hip/hip_bf16.h>

typedef __bf16 bf16;
typedef __attribute__((ext_vector_type(16))) __bf16 v16bf;
typedef __attribute__((ext_vector_type(8)))  float  v8f;
typedef __attribute__((ext_vector_type(4)))  unsigned int u32x4;
typedef __attribute__((ext_vector_type(4)))  float  f32x4;

#define NDIM  256
#define SEQ   512
#define BLK_M 128
#define BLK_K 32

// ---------------------------------------------------------------------------
// Kernel A: per-(b,s) precompute: position weight w[b,s], 1/(rowsum(adj)+1)
// one wave32 per adj row
// ---------------------------------------------------------------------------
__global__ __launch_bounds__(256) void prep_kernel(
    const float* __restrict__ adj, const int* __restrict__ text_len,
    const int* __restrict__ aspect_len, const int* __restrict__ left_len,
    float* __restrict__ w, float* __restrict__ dinv)
{
    int row  = blockIdx.x * 8 + (threadIdx.x >> 5);   // 0..65535
    int lane = threadIdx.x & 31;
    int b = row >> 9, s = row & (SEQ - 1);
    const float* arow = adj + (size_t)row * SEQ;
    float p = 0.f;
#pragma unroll
    for (int i = 0; i < 16; i++) p += arow[lane + i * 32];
#pragma unroll
    for (int off = 16; off > 0; off >>= 1) p += __shfl_xor(p, off, 32);

    int tl = text_len[b], al = aspect_len[b], st = left_len[b];
    int en = st + al - 1;
    float ctx = (float)(tl - al);
    float jf  = (float)s;
    float wt;
    if (s < st)       wt = 1.f - ((float)st - jf) / ctx;
    else if (s <= en) wt = 0.f;
    else if (s < tl)  wt = 1.f - (jf - (float)en) / ctx;
    else              wt = 0.f;

    if (lane == 0) { dinv[row] = 1.f / (p + 1.f); w[row] = wt; }
}

// ---------------------------------------------------------------------------
// Kernel G: W1,W2 -> transposed bf16 [N][K] so B fragments are K-contiguous
// ---------------------------------------------------------------------------
__global__ __launch_bounds__(256) void wprep_kernel(
    const float* __restrict__ W1, const float* __restrict__ W2,
    bf16* __restrict__ Wt1, bf16* __restrict__ Wt2)
{
    int idx = blockIdx.x * 256 + threadIdx.x;   // 0..65535
    int n = idx >> 8, k = idx & 255;
    Wt1[idx] = (bf16)W1[k * NDIM + n];
    Wt2[idx] = (bf16)W2[k * NDIM + n];
}

// ---------------------------------------------------------------------------
// Unified WMMA GEMM:  C[rows x 256] = A[rows x K] * Bt^T   (Bt is [256][K])
//   A_BF16  : A is bf16 (else fp32, optionally scaled by per-row wRow)
//   EPI_ROW : epilogue = relu(acc*dinv[r] + bias[d]) * wNext[r], bf16 row-major
//             else plain bf16 store transposed as out[b][d][s] (feeds next B)
//   PER_B   : Bt has a per-batch stride of 256*K (adj @ h case)
// 256 threads = 8 waves; each wave owns 16 M rows x all 16 N tiles.
// ---------------------------------------------------------------------------
template<bool A_BF16, bool EPI_ROW, bool PER_B, int K>
__global__ __launch_bounds__(256) void gemm_bf16_wmma(
    const void*  __restrict__ Aptr,
    const float* __restrict__ wRow,      // nullable pre-weight on A rows
    const bf16*  __restrict__ Bt,        // [N][K] (or per-batch)
    const float* __restrict__ dinv,
    const float* __restrict__ bias,
    const float* __restrict__ wNext,
    bf16* __restrict__ out)
{
    __shared__ bf16 Atile[BLK_M * BLK_K];
    __shared__ bf16 Btile[NDIM * BLK_K];

    const int tid   = threadIdx.x;
    const int wave  = tid >> 5, lane = tid & 31;
    const int l15   = lane & 15, lhalf = lane >> 4;
    const int Rbase = blockIdx.x * BLK_M;
    const int batch = Rbase / SEQ;
    const bf16* BtB = Bt + (PER_B ? (size_t)batch * NDIM * K : (size_t)0);

    v8f acc[16] = {};

    const int arow   = tid >> 1;           // 0..127
    const int achunk = (tid & 1) * 16;     // 0 or 16

    for (int k0 = 0; k0 < K; k0 += BLK_K) {
        // ---- stage A panel [128 x 32] (fp32*w -> bf16, or raw bf16) ----
        {
            const int r = Rbase + arow;
            bf16* dst = &Atile[arow * BLK_K + achunk];
            if (A_BF16) {
                const u32x4* src = (const u32x4*)((const bf16*)Aptr +
                                   (size_t)r * K + k0 + achunk);
                ((u32x4*)dst)[0] = src[0];
                ((u32x4*)dst)[1] = src[1];
            } else {
                const float wgt = wRow ? wRow[r] : 1.0f;
                const f32x4* src = (const f32x4*)((const float*)Aptr +
                                   (size_t)r * K + k0 + achunk);
#pragma unroll
                for (int q = 0; q < 4; q++) {
                    f32x4 f = src[q];
#pragma unroll
                    for (int e = 0; e < 4; e++)
                        dst[q * 4 + e] = (bf16)(f[e] * wgt);
                }
            }
        }
        // ---- stage B panel [256 x 32]: thread t copies row n=t ----
        {
            const bf16* src = BtB + (size_t)tid * K + k0;
            u32x4* dst = (u32x4*)&Btile[tid * BLK_K];
            dst[0] = ((const u32x4*)src)[0];
            dst[1] = ((const u32x4*)src)[1];
        }
        __syncthreads();

        // ---- A fragment: row M=l15 of this wave's subtile; K contiguous (k = 16*lhalf + s), same as B
        union { v16bf v; u32x4 u[2]; } af;
        {
            const bf16* ap = &Atile[(wave * 16 + l15) * BLK_K];
            af.u[0] = *(const u32x4*)(ap + lhalf * 16);
            af.u[1] = *(const u32x4*)(ap + lhalf * 16 + 8);
        }
#pragma unroll
        for (int nt = 0; nt < 16; nt++) {
            union { v16bf v; u32x4 u[2]; } bg;
            const bf16* bp = &Btile[(nt * 16 + l15) * BLK_K + lhalf * 16];
            bg.u[0] = ((const u32x4*)bp)[0];
            bg.u[1] = ((const u32x4*)bp)[1];
            acc[nt] = __builtin_amdgcn_wmma_f32_16x16x32_bf16(
                false, af.v, false, bg.v, (short)0, acc[nt], false, false);
        }
        __syncthreads();
    }

    // ---- epilogue: D layout lane=(l15,lhalf), elem e -> (M=e+8*lhalf, N=l15)
    const int m0 = wave * 16 + lhalf * 8;
    if (EPI_ROW) {
        float dv[8], wv[8];
#pragma unroll
        for (int e = 0; e < 8; e++) {
            int r = Rbase + m0 + e;
            dv[e] = dinv[r]; wv[e] = wNext[r];
        }
#pragma unroll
        for (int nt = 0; nt < 16; nt++) {
            int d = nt * 16 + l15;
            float bb = bias[d];
#pragma unroll
            for (int e = 0; e < 8; e++) {
                int r = Rbase + m0 + e;
                float val = acc[nt][e] * dv[e] + bb;
                val = val > 0.f ? val : 0.f;
                out[(size_t)r * NDIM + d] = (bf16)(val * wv[e]);
            }
        }
    } else {
#pragma unroll
        for (int nt = 0; nt < 16; nt++) {
            int d = nt * 16 + l15;
#pragma unroll
            for (int e = 0; e < 8; e++) {
                int r = Rbase + m0 + e;
                int b = r >> 9, s = r & (SEQ - 1);
                out[(size_t)b * NDIM * SEQ + (size_t)d * SEQ + s] =
                    (bf16)acc[nt][e];
            }
        }
    }
}

// ---------------------------------------------------------------------------
// Kernel E: only aspect rows of layer-2 adj-GEMM (<=16 rows per batch),
// fused relu + mask + reduction to v[b,d] = sum_s x[b,s,d].
// 1 WG per batch; wave w handles 32 N columns; 16 M rows via one WMMA tile.
// ---------------------------------------------------------------------------
__global__ __launch_bounds__(256) void aspect_kernel(
    const float* __restrict__ adj, const bf16* __restrict__ h2T,
    const float* __restrict__ dinv, const float* __restrict__ b2,
    const int* __restrict__ aspect_len, const int* __restrict__ left_len,
    float* __restrict__ vout)
{
    __shared__ bf16  Atile[16 * BLK_K];
    __shared__ bf16  Btile[NDIM * BLK_K];
    __shared__ float vsh[NDIM];

    const int b = blockIdx.x;
    const int tid  = threadIdx.x;
    const int wave = tid >> 5, lane = tid & 31;
    const int l15  = lane & 15, lhalf = lane >> 4;
    const int st = left_len[b], al = aspect_len[b];
    vsh[tid] = 0.f;

    v8f acc[2] = {};

    // A staging: 16x32 tile, 2 elems/thread
    const int arow = tid >> 4;
    const int acol = (tid & 15) * 2;
    int srow = st + arow; if (srow > SEQ - 1) srow = SEQ - 1;   // clamp pad rows
    const float* abase = adj + ((size_t)b * SEQ + srow) * SEQ;
    const bf16*  hb    = h2T + (size_t)b * NDIM * SEQ;

    for (int k0 = 0; k0 < SEQ; k0 += BLK_K) {
        Atile[arow * BLK_K + acol]     = (bf16)abase[k0 + acol];
        Atile[arow * BLK_K + acol + 1] = (bf16)abase[k0 + acol + 1];
        {
            const bf16* src = hb + (size_t)tid * SEQ + k0;
            u32x4* dst = (u32x4*)&Btile[tid * BLK_K];
            dst[0] = ((const u32x4*)src)[0];
            dst[1] = ((const u32x4*)src)[1];
        }
        __syncthreads();

        union { v16bf v; u32x4 u[2]; } af;
        const bf16* ap = &Atile[l15 * BLK_K];
        af.u[0] = *(const u32x4*)(ap + lhalf * 16);
        af.u[1] = *(const u32x4*)(ap + lhalf * 16 + 8);
#pragma unroll
        for (int nt = 0; nt < 2; nt++) {
            union { v16bf v; u32x4 u[2]; } bg;
            const bf16* bp = &Btile[(wave * 32 + nt * 16 + l15) * BLK_K + lhalf * 16];
            bg.u[0] = ((const u32x4*)bp)[0];
            bg.u[1] = ((const u32x4*)bp)[1];
            acc[nt] = __builtin_amdgcn_wmma_f32_16x16x32_bf16(
                false, af.v, false, bg.v, (short)0, acc[nt], false, false);
        }
        __syncthreads();
    }

    // relu + aspect mask + sum over rows -> v[b,d]
#pragma unroll
    for (int nt = 0; nt < 2; nt++) {
        int d = wave * 32 + nt * 16 + l15;
        float bb = b2[d];
        float sum = 0.f;
#pragma unroll
        for (int e = 0; e < 8; e++) {
            int i = e + lhalf * 8;            // row index within aspect span
            if (i < al) {
                int s = st + i;
                float val = acc[nt][e] * dinv[b * SEQ + s] + bb;
                sum += val > 0.f ? val : 0.f;
            }
        }
        atomicAdd(&vsh[d], sum);
    }
    __syncthreads();
    vout[b * NDIM + tid] = vsh[tid];
}

// ---------------------------------------------------------------------------
// Kernel F: scores = v . text_out[b,t,:], softmax over t, out = alpha @ text
// ---------------------------------------------------------------------------
__global__ __launch_bounds__(256) void attn_kernel(
    const float* __restrict__ text_out, const float* __restrict__ v,
    float* __restrict__ out)
{
    __shared__ float vsh[NDIM];
    __shared__ float sc[SEQ];
    __shared__ float red[256];

    const int b = blockIdx.x;
    const int tid = threadIdx.x;
    const int wave = tid >> 5, lane = tid & 31;
    vsh[tid] = v[b * NDIM + tid];
    __syncthreads();

    const float* tb = text_out + (size_t)b * SEQ * NDIM;
    for (int t = wave; t < SEQ; t += 8) {
        const float* row = tb + (size_t)t * NDIM;
        float p = 0.f;
#pragma unroll
        for (int j = 0; j < 8; j++) p += row[lane + j * 32] * vsh[lane + j * 32];
#pragma unroll
        for (int off = 16; off > 0; off >>= 1) p += __shfl_xor(p, off, 32);
        if (lane == 0) sc[t] = p;
    }
    __syncthreads();

    // softmax over 512 (threads cover tid, tid+256)
    red[tid] = fmaxf(sc[tid], sc[tid + 256]);
    __syncthreads();
    for (int off = 128; off > 0; off >>= 1) {
        if (tid < off) red[tid] = fmaxf(red[tid], red[tid + off]);
        __syncthreads();
    }
    float mx = red[0];
    __syncthreads();
    float e0 = __expf(sc[tid] - mx), e1 = __expf(sc[tid + 256] - mx);
    sc[tid] = e0; sc[tid + 256] = e1;
    red[tid] = e0 + e1;
    __syncthreads();
    for (int off = 128; off > 0; off >>= 1) {
        if (tid < off) red[tid] += red[tid + off];
        __syncthreads();
    }
    float inv = 1.f / red[0];
    __syncthreads();

    // out[b,d] = sum_t alpha[t] * text_out[b,t,d]; thread = d (coalesced)
    float a = 0.f;
    for (int t = 0; t < SEQ; t++) a += sc[t] * tb[(size_t)t * NDIM + tid];
    out[b * NDIM + tid] = a * inv;
}

// ---------------------------------------------------------------------------
extern "C" void kernel_launch(void* const* d_in, const int* in_sizes, int n_in,
                              void* d_out, int out_size, void* d_ws, size_t ws_size,
                              hipStream_t stream)
{
    const float* text_out  = (const float*)d_in[0];
    const float* adj       = (const float*)d_in[1];
    const float* W1        = (const float*)d_in[2];
    const float* b1        = (const float*)d_in[3];
    const float* W2        = (const float*)d_in[4];
    const float* b2        = (const float*)d_in[5];
    const int*   text_len  = (const int*)d_in[6];
    const int*   aspect_ln = (const int*)d_in[7];
    const int*   left_len  = (const int*)d_in[8];
    float*       out       = (float*)d_out;

    char* ws = (char*)d_ws;
    float* w    = (float*)(ws + 0x00000);                     // 256 KB
    float* dinv = (float*)(ws + 0x40000);                     // 256 KB
    bf16*  Wt1  = (bf16*) (ws + 0x80000);                     // 128 KB
    bf16*  Wt2  = (bf16*) (ws + 0xA0000);                     // 128 KB
    float* v    = (float*)(ws + 0xC0000);                     // 128 KB
    bf16*  hT   = (bf16*) (ws + 0xE0000);                     // 32 MB (reused as h2T)
    bf16*  z    = (bf16*) (ws + 0xE0000 + (32u << 20));       // 32 MB

    // per-(b,s) scalars
    prep_kernel<<<8192, 256, 0, stream>>>(adj, text_len, aspect_ln, left_len, w, dinv);
    // weight transpose + bf16
    wprep_kernel<<<256, 256, 0, stream>>>(W1, W2, Wt1, Wt2);
    // hT[b][d][s] = (w .* text_out) @ W1
    gemm_bf16_wmma<false, false, false, 256><<<512, 256, 0, stream>>>(
        (const void*)text_out, w, Wt1, nullptr, nullptr, nullptr, hT);
    // z[r][d] = w .* relu(adj @ h / denom + b1)
    gemm_bf16_wmma<false, true, true, 512><<<512, 256, 0, stream>>>(
        (const void*)adj, nullptr, hT, dinv, b1, w, z);
    // h2T[b][d][s] = z @ W2   (reuse hT buffer)
    gemm_bf16_wmma<true, false, false, 256><<<512, 256, 0, stream>>>(
        (const void*)z, nullptr, Wt2, nullptr, nullptr, nullptr, hT);
    // aspect-row adj GEMM + relu + mask + reduce -> v[b][d]
    aspect_kernel<<<128, 256, 0, stream>>>(adj, hT, dinv, b2, aspect_ln, left_len, v);
    // softmax attention -> out
    attn_kernel<<<128, 256, 0, stream>>>(text_out, v, out);
}
